// _DenseLayer_60335700574774
// MI455X (gfx1250) — 8-bit matrix operand order repaired and checked by reading, compiled, not run on hardware
//
#include <hip/hip_runtime.h>
#include <math.h>

typedef int i32x8 __attribute__((ext_vector_type(8)));

constexpr int BN   = 64;
constexpr int CIN  = 512;
constexpr int P    = 784;
constexpr int CMID = 128;
constexpr int COUT = 32;
constexpr int CCAT = 544;
constexpr int NPIX = BN * P;
constexpr float EPSF = 1e-5f;

constexpr size_t OFF_SLOTS     = 0;
constexpr size_t OFF_BN1_MEAN  = 4096;
constexpr size_t OFF_BN1_RSTD  = 6144;
constexpr size_t OFF_RBN_CMAX  = 8192;
constexpr size_t OFF_RBN_CMIN  = 16384;
constexpr size_t OFF_RBN_CSUM  = 24576;
constexpr size_t OFF_RBN_MEAN  = 32768;
constexpr size_t OFF_RBN_SCALE = 33280;
constexpr size_t OFF_SUMQW1    = 36864;
constexpr size_t OFF_SUMQW2    = 40960;
constexpr size_t OFF_QW1       = 45056;
constexpr size_t OFF_QW2T      = 110592;
constexpr size_t OFF_COLSUM1   = 147456;
constexpr size_t OFF_COLSUM3   = 348160;
constexpr size_t OFF_QX1       = 548864;
constexpr size_t OFF_QX3       = 26238976;
constexpr size_t OFF_H2        = 32661504;

__device__ __forceinline__ unsigned enc_f(float f) {
  unsigned u = __float_as_uint(f);
  return (u >> 31) ? ~u : (u | 0x80000000u);
}
__device__ __forceinline__ float dec_f(unsigned e) {
  return __uint_as_float((e >> 31) ? (e & 0x7fffffffu) : ~e);
}
__device__ __forceinline__ float qsc(float mn, float mx) {
  return fmaxf((mx - mn) * (1.0f / 255.0f), 1e-8f);
}
__device__ __forceinline__ int qidx(float v, float mn, float mx, float s) {
  return (int)rintf((fminf(fmaxf(v, mn), mx) - mn) / s);
}
__device__ __forceinline__ float qdq(float v, float mn, float mx) {
  float s = qsc(mn, mx);
  return (float)qidx(v, mn, mx, s) * s + mn;
}

__global__ void k_init(unsigned* slots) {
  int t = threadIdx.x;
  if (t < 12) slots[t] = (t & 1) ? 0u : 0xFFFFFFFFu;
}

__global__ __launch_bounds__(256) void k_bn1_stats(const float* x, float* mean, float* rstd) {
  int c = blockIdx.x, tid = threadIdx.x;
  float s = 0.f, s2 = 0.f;
  for (int i = tid; i < NPIX; i += 256) {
    int b = i / P, p = i % P;
    float v = x[((size_t)b * CIN + c) * P + p];
    s += v; s2 += v * v;
  }
  __shared__ float sh1[256], sh2[256];
  sh1[tid] = s; sh2[tid] = s2; __syncthreads();
  for (int w = 128; w > 0; w >>= 1) {
    if (tid < w) { sh1[tid] += sh1[tid + w]; sh2[tid] += sh2[tid + w]; }
    __syncthreads();
  }
  if (tid == 0) {
    float m = sh1[0] / (float)NPIX;
    float var = sh2[0] / (float)NPIX - m * m;
    mean[c] = m;
    rstd[c] = rsqrtf(var + EPSF);
  }
}

__global__ void k_copy_x(const float* x, float* out) {
  size_t total = (size_t)BN * CIN * P;
  for (size_t i = (size_t)blockIdx.x * blockDim.x + threadIdx.x; i < total;
       i += (size_t)gridDim.x * blockDim.x) {
    int p = (int)(i % P); size_t t = i / P;
    int c = (int)(t % CIN); int b = (int)(t / CIN);
    out[((size_t)b * CCAT + c) * P + p] = x[i];
  }
}

__global__ __launch_bounds__(256) void k_minmax(const float* v, int n, unsigned* slots, int idx) {
  int tid = threadIdx.x;
  float lmn = 3.4e38f, lmx = -3.4e38f;
  for (int i = blockIdx.x * 256 + tid; i < n; i += gridDim.x * 256) {
    float f = v[i]; lmn = fminf(lmn, f); lmx = fmaxf(lmx, f);
  }
  __shared__ float smn[256], smx[256];
  smn[tid] = lmn; smx[tid] = lmx; __syncthreads();
  for (int w = 128; w > 0; w >>= 1) {
    if (tid < w) { smn[tid] = fminf(smn[tid], smn[tid + w]); smx[tid] = fmaxf(smx[tid], smx[tid + w]); }
    __syncthreads();
  }
  if (tid == 0) { atomicMin(&slots[idx], enc_f(smn[0])); atomicMax(&slots[idx + 1], enc_f(smx[0])); }
}

__global__ __launch_bounds__(256) void k_h1_minmax(const float* x, const float* mean, const float* rstd,
                                                   const float* w, const float* b, unsigned* slots) {
  int tid = threadIdx.x;
  size_t total = (size_t)BN * CIN * P;
  float lmn = 3.4e38f, lmx = -3.4e38f;
  for (size_t i = (size_t)blockIdx.x * 256 + tid; i < total; i += (size_t)gridDim.x * 256) {
    int c = (int)((i / P) % CIN);
    float h = fmaxf((x[i] - mean[c]) * rstd[c] * w[c] + b[c], 0.f);
    lmn = fminf(lmn, h); lmx = fmaxf(lmx, h);
  }
  __shared__ float smn[256], smx[256];
  smn[tid] = lmn; smx[tid] = lmx; __syncthreads();
  for (int w2 = 128; w2 > 0; w2 >>= 1) {
    if (tid < w2) { smn[tid] = fminf(smn[tid], smn[tid + w2]); smx[tid] = fmaxf(smx[tid], smx[tid + w2]); }
    __syncthreads();
  }
  if (tid == 0) { atomicMin(&slots[0], enc_f(smn[0])); atomicMax(&slots[1], enc_f(smx[0])); }
}

__global__ __launch_bounds__(256) void k_quant_x1(const float* x, const float* mean, const float* rstd,
                                                  const float* w, const float* b, const unsigned* slots,
                                                  unsigned char* qx1) {
  float mn = dec_f(slots[0]), mx = dec_f(slots[1]);
  float s = qsc(mn, mx);
  size_t total = (size_t)BN * CIN * P;
  for (size_t i = (size_t)blockIdx.x * blockDim.x + threadIdx.x; i < total;
       i += (size_t)gridDim.x * blockDim.x) {
    int p = (int)(i % P); size_t t = i / P;
    int c = (int)(t % CIN); int bb = (int)(t / CIN);
    float h = fmaxf((x[i] - mean[c]) * rstd[c] * w[c] + b[c], 0.f);
    qx1[((size_t)bb * P + p) * CIN + c] = (unsigned char)qidx(h, mn, mx, s);
  }
}

__global__ void k_colsum(const unsigned char* q, int* out, int C) {
  int idx = blockIdx.x * blockDim.x + threadIdx.x;
  if (idx >= NPIX) return;
  const unsigned* p32 = (const unsigned*)(q + (size_t)idx * C);
  int s = 0;
  for (int j = 0; j < C / 4; ++j) {
    unsigned u = p32[j];
    s += (int)(u & 255u) + (int)((u >> 8) & 255u) + (int)((u >> 16) & 255u) + (int)(u >> 24);
  }
  out[idx] = s;
}

__global__ __launch_bounds__(256) void k_quant_w1(const float* w1, const unsigned* slots,
                                                  unsigned char* qw1, int* sumqw1) {
  int m = blockIdx.x, tid = threadIdx.x;
  float mn = dec_f(slots[2]), mx = dec_f(slots[3]);
  float s = qsc(mn, mx);
  int ls = 0;
  for (int k = tid; k < CIN; k += 256) {
    int qi = qidx(w1[(size_t)m * CIN + k], mn, mx, s);
    qw1[(size_t)m * CIN + k] = (unsigned char)qi;
    ls += qi;
  }
  __shared__ int sh[256];
  sh[tid] = ls; __syncthreads();
  for (int w = 128; w > 0; w >>= 1) { if (tid < w) sh[tid] += sh[tid + w]; __syncthreads(); }
  if (tid == 0) sumqw1[m] = sh[0];
}

constexpr int BSTRIDE = 528;
__global__ __launch_bounds__(256) void k_gemm1(const unsigned char* __restrict__ qx1,
                                               const unsigned char* __restrict__ qw1,
                                               const int* __restrict__ colsum1,
                                               const int* __restrict__ sumqw1,
                                               const unsigned* __restrict__ slots,
                                               float* __restrict__ h2) {
  __shared__ unsigned char btile[16 * BSTRIDE];
  int tid = threadIdx.x;
  int b  = blockIdx.x / 49;
  int nt = blockIdx.x % 49;

  const unsigned char* gbase = qx1 + ((size_t)b * P + nt * 16) * CIN;
#pragma unroll
  for (int it = 0; it < 2; ++it) {
    int j   = tid + 256 * it;
    int col = j >> 5;
    int k16 = j & 31;
    unsigned lds_off = (unsigned)(size_t)(btile + col * BSTRIDE + k16 * 16);
    const unsigned char* g = gbase + (size_t)col * CIN + k16 * 16;
    asm volatile("global_load_async_to_lds_b128 %0, %1, off"
                 :: "v"(lds_off), "v"(g) : "memory");
  }
  asm volatile("s_wait_asynccnt 0x0" ::: "memory");
  __syncthreads();

  int mt   = tid >> 5;
  int lane = tid & 31;
  int hi = lane >> 4, l16 = lane & 15;
  int abase = hi * 8, bbofs = hi * 16;
  const unsigned char* arow   = qw1 + (size_t)(mt * 16 + l16) * CIN;
  const unsigned char* bcol_l = btile + l16 * BSTRIDE;

  i32x8 acc = {0, 0, 0, 0, 0, 0, 0, 0};
#pragma unroll
  for (int ks = 0; ks < CIN; ks += 64) {
    i32x8 A, Bf;
#pragma unroll
    for (int g = 0; g < 4; ++g) {
      const int* pa = (const int*)(arow + ks + 2 * abase + 8 * g + 16 * (g >> 1));
      A[2 * g] = pa[0]; A[2 * g + 1] = pa[1];
    }
    int4 b0 = *(const int4*)(bcol_l + ks + bbofs);
    int4 b1 = *(const int4*)(bcol_l + ks + bbofs + 32);
    Bf[0] = b0.x; Bf[1] = b0.y; Bf[2] = b0.z; Bf[3] = b0.w;
    Bf[4] = b1.x; Bf[5] = b1.y; Bf[6] = b1.z; Bf[7] = b1.w;
    acc = __builtin_amdgcn_wmma_i32_16x16x64_iu8(false, A, false, Bf, acc, false, false);
  }
  float mnx = dec_f(slots[0]), mxx = dec_f(slots[1]);
  float mnw = dec_f(slots[2]), mxw = dec_f(slots[3]);
  float sx = qsc(mnx, mxx), sw = qsc(mnw, mxw);
  int p = nt * 16 + l16;
  float cterm = mnw * sx * (float)colsum1[b * P + p] + (float)CIN * mnx * mnw;
#pragma unroll
  for (int rr = 0; rr < 8; ++rr) {
    int m = mt * 16 + rr + hi * 8;
    float val = sx * sw * (float)acc[rr] + mnx * sw * (float)sumqw1[m] + cterm;
    h2[((size_t)b * CMID + m) * P + p] = val;
  }
}

__global__ __launch_bounds__(256) void k_rbn_stats(const float* h2, float* cmax, float* cmin, float* csum) {
  int c = blockIdx.x >> 4, k = blockIdx.x & 15, tid = threadIdx.x;
  float mx = -3.4e38f, mn = 3.4e38f, s = 0.f;
  for (int i = tid; i < 3136; i += 256) {
    int b = 4 * k + i / P, p = i % P;
    float v = h2[((size_t)b * CMID + c) * P + p];
    mx = fmaxf(mx, v); mn = fminf(mn, v); s += v;
  }
  __shared__ float sx[256], sn[256], ss[256];
  sx[tid] = mx; sn[tid] = mn; ss[tid] = s; __syncthreads();
  for (int w = 128; w > 0; w >>= 1) {
    if (tid < w) { sx[tid] = fmaxf(sx[tid], sx[tid + w]); sn[tid] = fminf(sn[tid], sn[tid + w]); ss[tid] += ss[tid + w]; }
    __syncthreads();
  }
  if (tid == 0) { cmax[c * 16 + k] = sx[0]; cmin[c * 16 + k] = sn[0]; csum[c * 16 + k] = ss[0]; }
}

__global__ void k_rbn_final(const float* cmax, const float* cmin, const float* csum,
                            float* rmean, float* rscale) {
  int c = threadIdx.x;
  if (c >= CMID) return;
  float mmax = 0.f, mmin = 0.f, tot = 0.f;
  for (int k = 0; k < 16; ++k) { mmax += cmax[c * 16 + k]; mmin += cmin[c * 16 + k]; tot += csum[c * 16 + k]; }
  mmax *= (1.0f / 16.0f); mmin *= (1.0f / 16.0f);
  double sf = 0.175 * (1.0 + sqrt(M_PI * log(4.0))) / sqrt(2.0 * log(3136.0));
  rmean[c] = tot / (float)NPIX;
  rscale[c] = 1.0f / ((mmax - mmin) * (float)sf + EPSF);
}

__device__ __forceinline__ float rbn_relu(float v, int c, const float* rmean, const float* rscale,
                                          const float* rw, const float* rb, const unsigned* slots) {
  float qw = qdq(rw[c], dec_f(slots[8]), dec_f(slots[9]));
  float qb = qdq(rb[c], dec_f(slots[10]), dec_f(slots[11]));
  return fmaxf((v - rmean[c]) * rscale[c] * qw + qb, 0.f);
}

__global__ __launch_bounds__(256) void k_h3_minmax(const float* h2, const float* rmean, const float* rscale,
                                                   const float* rw, const float* rb, unsigned* slots) {
  int tid = threadIdx.x;
  size_t total = (size_t)BN * CMID * P;
  float lmn = 3.4e38f, lmx = -3.4e38f;
  for (size_t i = (size_t)blockIdx.x * 256 + tid; i < total; i += (size_t)gridDim.x * 256) {
    int c = (int)((i / P) % CMID);
    float h = rbn_relu(h2[i], c, rmean, rscale, rw, rb, slots);
    lmn = fminf(lmn, h); lmx = fmaxf(lmx, h);
  }
  __shared__ float smn[256], smx[256];
  smn[tid] = lmn; smx[tid] = lmx; __syncthreads();
  for (int w = 128; w > 0; w >>= 1) {
    if (tid < w) { smn[tid] = fminf(smn[tid], smn[tid + w]); smx[tid] = fmaxf(smx[tid], smx[tid + w]); }
    __syncthreads();
  }
  if (tid == 0) { atomicMin(&slots[4], enc_f(smn[0])); atomicMax(&slots[5], enc_f(smx[0])); }
}

__global__ __launch_bounds__(256) void k_quant_x3(const float* h2, const float* rmean, const float* rscale,
                                                  const float* rw, const float* rb, const unsigned* slots,
                                                  unsigned char* qx3) {
  float mn = dec_f(slots[4]), mx = dec_f(slots[5]);
  float s = qsc(mn, mx);
  size_t total = (size_t)BN * CMID * P;
  for (size_t i = (size_t)blockIdx.x * blockDim.x + threadIdx.x; i < total;
       i += (size_t)gridDim.x * blockDim.x) {
    int p = (int)(i % P); size_t t = i / P;
    int c = (int)(t % CMID); int b = (int)(t / CMID);
    float h = rbn_relu(h2[i], c, rmean, rscale, rw, rb, slots);
    qx3[((size_t)b * P + p) * CMID + c] = (unsigned char)qidx(h, mn, mx, s);
  }
}

__global__ __launch_bounds__(128) void k_quant_w2(const float* w2, const unsigned* slots,
                                                  unsigned char* qw2t, int* sumqw2) {
  int t = blockIdx.x / COUT, m = blockIdx.x % COUT, c = threadIdx.x;
  float mn = dec_f(slots[6]), mx = dec_f(slots[7]);
  float s = qsc(mn, mx);
  int qi = qidx(w2[((size_t)m * CMID + c) * 9 + t], mn, mx, s);
  qw2t[((size_t)t * COUT + m) * CMID + c] = (unsigned char)qi;
  __shared__ int sh[128];
  sh[c] = qi; __syncthreads();
  for (int w = 64; w > 0; w >>= 1) { if (c < w) sh[c] += sh[c + w]; __syncthreads(); }
  if (c == 0) sumqw2[t * COUT + m] = sh[0];
}

__global__ __launch_bounds__(256) void k_gemm2(const unsigned char* __restrict__ qx3,
                                               const unsigned char* __restrict__ qw2t,
                                               const int* __restrict__ colsum3,
                                               const int* __restrict__ sumqw2,
                                               const unsigned* __restrict__ slots,
                                               float* __restrict__ out) {
  int wave = (blockIdx.x << 3) + (threadIdx.x >> 5);
  int lane = threadIdx.x & 31;
  int b  = wave / (2 * 49);
  int r  = wave % (2 * 49);
  int mt = r / 49;
  int nt = r % 49;
  int hi = lane >> 4, l16 = lane & 15;
  int abase = hi * 8, bbase = hi * 16;
  int p = nt * 16 + l16;
  int y = p / 28, x = p % 28;

  i32x8 acc = {0, 0, 0, 0, 0, 0, 0, 0};
  unsigned vmask = 0;
  int sqx = 0, cnt = 0;
  for (int t = 0; t < 9; ++t) {
    int dy = t / 3 - 1, dx = t % 3 - 1;
    int yy = y + dy, xx = x + dx;
    bool valid = (yy >= 0) && (yy < 28) && (xx >= 0) && (xx < 28);
    int pp = valid ? (yy * 28 + xx) : 0;
    if (valid) { vmask |= (1u << t); ++cnt; sqx += colsum3[b * P + pp]; }
    const unsigned char* arow = qw2t + ((size_t)t * COUT + mt * 16 + l16) * CMID;
    const unsigned char* bcol = qx3 + ((size_t)b * P + pp) * CMID;
#pragma unroll
    for (int ks = 0; ks < CMID; ks += 64) {
      i32x8 A, Bf;
#pragma unroll
      for (int g = 0; g < 4; ++g) {
        const int* pa = (const int*)(arow + ks + 2 * abase + 8 * g + 16 * (g >> 1));
        A[2 * g] = pa[0]; A[2 * g + 1] = pa[1];
      }
      if (valid) {
        int4 b0 = *(const int4*)(bcol + ks + bbase);
        int4 b1 = *(const int4*)(bcol + ks + bbase + 32);
        Bf[0] = b0.x; Bf[1] = b0.y; Bf[2] = b0.z; Bf[3] = b0.w;
        Bf[4] = b1.x; Bf[5] = b1.y; Bf[6] = b1.z; Bf[7] = b1.w;
      } else {
        Bf = (i32x8){0, 0, 0, 0, 0, 0, 0, 0};
      }
      acc = __builtin_amdgcn_wmma_i32_16x16x64_iu8(false, A, false, Bf, acc, false, false);
    }
  }
  float mnx = dec_f(slots[4]), mxx = dec_f(slots[5]);
  float mnw = dec_f(slots[6]), mxw = dec_f(slots[7]);
  float sx = qsc(mnx, mxx), sw = qsc(mnw, mxw);
  float cterm = mnw * sx * (float)sqx + mnx * mnw * (float)(CMID * cnt);
#pragma unroll
  for (int rr = 0; rr < 8; ++rr) {
    int m = mt * 16 + rr + hi * 8;
    int sqw = 0;
    for (int t = 0; t < 9; ++t)
      if ((vmask >> t) & 1u) sqw += sumqw2[t * COUT + m];
    float val = sx * sw * (float)acc[rr] + mnx * sw * (float)sqw + cterm;
    out[((size_t)b * CCAT + 512 + m) * P + p] = val;
  }
}

extern "C" void kernel_launch(void* const* d_in, const int* in_sizes, int n_in,
                              void* d_out, int out_size, void* d_ws, size_t ws_size,
                              hipStream_t stream) {
  (void)in_sizes; (void)n_in; (void)out_size; (void)ws_size;
  const float* x    = (const float*)d_in[0];
  const float* bn1w = (const float*)d_in[1];
  const float* bn1b = (const float*)d_in[2];
  const float* w1   = (const float*)d_in[3];
  const float* rbnw = (const float*)d_in[4];
  const float* rbnb = (const float*)d_in[5];
  const float* w2   = (const float*)d_in[6];
  float* out = (float*)d_out;
  char* ws = (char*)d_ws;

  unsigned* slots  = (unsigned*)(ws + OFF_SLOTS);
  float* bn1_mean  = (float*)(ws + OFF_BN1_MEAN);
  float* bn1_rstd  = (float*)(ws + OFF_BN1_RSTD);
  float* cmax      = (float*)(ws + OFF_RBN_CMAX);
  float* cmin      = (float*)(ws + OFF_RBN_CMIN);
  float* csum      = (float*)(ws + OFF_RBN_CSUM);
  float* rmean     = (float*)(ws + OFF_RBN_MEAN);
  float* rscale    = (float*)(ws + OFF_RBN_SCALE);
  int* sumqw1      = (int*)(ws + OFF_SUMQW1);
  int* sumqw2      = (int*)(ws + OFF_SUMQW2);
  unsigned char* qw1  = (unsigned char*)(ws + OFF_QW1);
  unsigned char* qw2t = (unsigned char*)(ws + OFF_QW2T);
  int* colsum1     = (int*)(ws + OFF_COLSUM1);
  int* colsum3     = (int*)(ws + OFF_COLSUM3);
  unsigned char* qx1 = (unsigned char*)(ws + OFF_QX1);
  unsigned char* qx3 = (unsigned char*)(ws + OFF_QX3);
  float* h2        = (float*)(ws + OFF_H2);

  k_init<<<1, 32, 0, stream>>>(slots);
  k_bn1_stats<<<CIN, 256, 0, stream>>>(x, bn1_mean, bn1_rstd);
  k_copy_x<<<4096, 256, 0, stream>>>(x, out);
  k_minmax<<<64, 256, 0, stream>>>(w1, CMID * CIN, slots, 2);
  k_minmax<<<1, 256, 0, stream>>>(rbnw, CMID, slots, 8);
  k_minmax<<<1, 256, 0, stream>>>(rbnb, CMID, slots, 10);
  k_minmax<<<36, 256, 0, stream>>>(w2, COUT * CMID * 9, slots, 6);
  k_h1_minmax<<<2048, 256, 0, stream>>>(x, bn1_mean, bn1_rstd, bn1w, bn1b, slots);
  k_quant_x1<<<4096, 256, 0, stream>>>(x, bn1_mean, bn1_rstd, bn1w, bn1b, slots, qx1);
  k_colsum<<<(NPIX + 255) / 256, 256, 0, stream>>>(qx1, colsum1, CIN);
  k_quant_w1<<<CMID, 256, 0, stream>>>(w1, slots, qw1, sumqw1);
  k_gemm1<<<64 * 49, 256, 0, stream>>>(qx1, qw1, colsum1, sumqw1, slots, h2);
  k_rbn_stats<<<CMID * 16, 256, 0, stream>>>(h2, cmax, cmin, csum);
  k_rbn_final<<<1, 128, 0, stream>>>(cmax, cmin, csum, rmean, rscale);
  k_h3_minmax<<<1024, 256, 0, stream>>>(h2, rmean, rscale, rbnw, rbnb, slots);
  k_quant_x3<<<1024, 256, 0, stream>>>(h2, rmean, rscale, rbnw, rbnb, slots, qx3);
  k_colsum<<<(NPIX + 255) / 256, 256, 0, stream>>>(qx3, colsum3, CMID);
  k_quant_w2<<<9 * COUT, 128, 0, stream>>>(w2, slots, qw2t, sumqw2);
  k_gemm2<<<(64 * 2 * 49) / 8, 256, 0, stream>>>(qx3, qw2t, colsum3, sumqw2, slots, out);
}
